// NeuralRenderer_31421980737964
// MI455X (gfx1250) — matrix operand order repaired and checked by reading, not compiled, not run on hardware
//
#include <hip/hip_runtime.h>
#include <hip/hip_bf16.h>
#include <math.h>

typedef __attribute__((ext_vector_type(16))) _Float16 v16h;
typedef __attribute__((ext_vector_type(8)))  _Float16 v8h;
typedef __attribute__((ext_vector_type(8)))  float    v8f;

#define N_LEVELS 16
#define TBL      (1 << 19)
#define NS       128
#define NEARZ    2.0f
#define FARZ     6.0f
#define AS       72   /* padded f16 stride of activation buffer: 144 B, 16 B aligned */

#define WMMA(a, b, c) \
  __builtin_amdgcn_wmma_f32_16x16x32_f16(false, (a), false, (b), (short)0, (c), false, false)

// ---- A fragment: 16x32 f16. ISA 7.12.2: lane L (row = L&15, hi = L>>4) holds
// two contiguous runs: K = hi*8..+7 and K = 16+hi*8..+7 -> two 16-byte LDS loads.
__device__ __forceinline__ v16h frag_a(const _Float16* __restrict__ src, int lane, int stride) {
  int row = lane & 15, hi = lane >> 4;
  const _Float16* p = src + row * stride + hi * 8;
  union { v16h v; v8h h[2]; } u;
  u.h[0] = *(const v8h*)p;
  u.h[1] = *(const v8h*)(p + 16);
  return u.v;
}

// ---- B fragment from pre-swizzled weights: one aligned 32-byte v16h load per lane.
// Swizzle: chunk = ((kb*nblk + nb)*32 + hi*16 + col), hi = (k>>3)&1, pos = ((k>>4)&1)*8 + (k&7)
// (lane hi covers K = kb*32 + hi*8 .. +7 and kb*32 + 16 + hi*8 .. +7, column nb*16 + col).
__device__ __forceinline__ int bswz(int k, int n, int nblk) {
  return (((k >> 5) * nblk + (n >> 4)) * 32 + ((k >> 3) & 1) * 16 + (n & 15)) * 16 + ((k >> 4) & 1) * 8 + (k & 7);
}
__device__ __forceinline__ v16h frag_b(const _Float16* __restrict__ base, int lane,
                                       int kb, int nb, int nblk) {
  return *((const v16h*)base + (kb * nblk + nb) * 32 + lane);
}

// LDS layout (bytes):
//  wbuf   : 22528  (sigma phase: ws0[2048 f16]|ws1[4096]|ws2[1024]; color: wc0|wc1|wc2|wc3)
//  enchc  :  8192  (enc features [128][32] f16, reused as hc [128][32])
//  act    : 18432  ([128][AS] f16)
//  sig/keep/colb/alphab/shDir : f32 scratch
#define SM_WBUF   0
#define SM_ENCHC  22528
#define SM_ACT    30720
#define SM_SIG    49152
#define SM_KEEP   49664
#define SM_COL    50176
#define SM_ALPHA  51712
#define SM_SH     52224
#define SM_TOTAL  52288

__global__ __launch_bounds__(256)
void nerf_fused_wmma(const float* __restrict__ rays,
                     const float* __restrict__ tabs,
                     const float* __restrict__ ws0, const float* __restrict__ ws1,
                     const float* __restrict__ ws2,
                     const float* __restrict__ wc0, const float* __restrict__ wc1,
                     const float* __restrict__ wc2, const float* __restrict__ wc3,
                     float* __restrict__ out, int n_rays) {
  __shared__ __align__(32) unsigned char smem[SM_TOTAL];
  _Float16* wbuf  = (_Float16*)(smem + SM_WBUF);
  _Float16* enchc = (_Float16*)(smem + SM_ENCHC);
  _Float16* act   = (_Float16*)(smem + SM_ACT);
  float*    sig   = (float*)(smem + SM_SIG);
  float*    keep  = (float*)(smem + SM_KEEP);
  float*    colb  = (float*)(smem + SM_COL);
  float*    alphab= (float*)(smem + SM_ALPHA);
  float*    shDir = (float*)(smem + SM_SH);

  const int tid = threadIdx.x;
  const int ray = blockIdx.x;

  // ---- Phase 0: sigma-MLP weights f32 -> f16, written in B-fragment order ----
  for (int i = tid; i < 32 * 64; i += 256) { int k = i >> 6, n = i & 63; wbuf[bswz(k, n, 4)] = (_Float16)ws0[i]; }
  for (int i = tid; i < 64 * 64; i += 256) { int k = i >> 6, n = i & 63; wbuf[2048 + bswz(k, n, 4)] = (_Float16)ws1[i]; }
  for (int i = tid; i < 64 * 16; i += 256) { int k = i >> 4, n = i & 15; wbuf[6144 + bswz(k, n, 1)] = (_Float16)ws2[i]; }

  // ---- Phase 1: hash-grid encode (2 threads per sample, 8 levels each) ----
  const float ox = rays[ray * 6 + 0], oy = rays[ray * 6 + 1], oz = rays[ray * 6 + 2];
  const float dx = rays[ray * 6 + 3], dy = rays[ray * 6 + 4], dz = rays[ray * 6 + 5];
  {
    int s = tid >> 1, half = tid & 1;
    float z = NEARZ + (FARZ - NEARZ) * (float)s / (float)(NS - 1);
    float px = ox + dx * z, py = oy + dy * z, pz = oz + dz * z;
    float cx = fminf(fmaxf(px, -1.5f), 1.5f);
    float cy = fminf(fmaxf(py, -1.5f), 1.5f);
    float cz = fminf(fmaxf(pz, -1.5f), 1.5f);
    if (half == 0) keep[s] = (px == cx && py == cy && pz == cz) ? 1.f : 0.f;

    for (int l = half * 8; l < half * 8 + 8; ++l) {
      float res = floorf(16.f * exp2f((float)l * (1.f / 3.f)));  // floor(16 * 2^(l/3))
      float g = 3.f / res;
      int bx = (int)floorf((cx + 1.5f) / g);
      int by = (int)floorf((cy + 1.5f) / g);
      int bz = (int)floorf((cz + 1.5f) / g);
      float wx = (cx - ((float)bx * g - 1.5f)) / g;
      float wy = (cy - ((float)by * g - 1.5f)) / g;
      float wz = (cz - ((float)bz * g - 1.5f)) / g;
      float f0 = 0.f, f1 = 0.f;
#pragma unroll
      for (int c = 0; c < 8; ++c) {
        int ci = (c >> 2) & 1, cj = (c >> 1) & 1, ck = c & 1;
        unsigned ux = (unsigned)(bx + ci), uy = (unsigned)(by + cj), uz = (unsigned)(bz + ck);
        unsigned h = (ux * 1u) ^ (uy * 2654435761u) ^ (uz * 805459861u);
        const float* e = tabs + ((size_t)l * TBL + (size_t)(h & (TBL - 1))) * 2;
        float wt = (ci ? wx : 1.f - wx) * (cj ? wy : 1.f - wy) * (ck ? wz : 1.f - wz);
        f0 += e[0] * wt; f1 += e[1] * wt;
      }
      enchc[s * 32 + 2 * l]     = (_Float16)f0;
      enchc[s * 32 + 2 * l + 1] = (_Float16)f1;
    }
  }
  if (tid == 0) {  // SH deg-4 basis: direction constant along the ray
    float xx = dx * dx, yy = dy * dy, zz = dz * dz;
    float xy = dx * dy, yz = dy * dz, xz = dx * dz;
    shDir[0]  = 0.28209479177387814f;
    shDir[1]  = -0.4886025119029199f * dy;
    shDir[2]  =  0.4886025119029199f * dz;
    shDir[3]  = -0.4886025119029199f * dx;
    shDir[4]  =  1.0925484305920792f * xy;
    shDir[5]  = -1.0925484305920792f * yz;
    shDir[6]  =  0.31539156525252005f * (2.f * zz - xx - yy);
    shDir[7]  = -1.0925484305920792f * xz;
    shDir[8]  =  0.5462742152960396f * (xx - yy);
    shDir[9]  = -0.5900435899266435f * dy * (3.f * xx - yy);
    shDir[10] =  2.890611442640554f  * xy * dz;
    shDir[11] = -0.4570457994644658f * dy * (4.f * zz - xx - yy);
    shDir[12] =  0.3731763325901154f * dz * (2.f * zz - 3.f * xx - 3.f * yy);
    shDir[13] = -0.4570457994644658f * dx * (4.f * zz - xx - yy);
    shDir[14] =  1.445305721320277f  * dz * (xx - yy);
    shDir[15] = -0.5900435899266435f * dx * (xx - 3.f * yy);
  }
  __syncthreads();

  const int lane = tid & 31, wv = tid >> 5;
  const int s0 = wv * 16;
  const int col = lane & 15, hi = lane >> 4;

  // ---- Phase 2: sigma MLP 32->64->64->16 ----
  {
    const _Float16* ws0h = wbuf;
    const _Float16* ws1h = wbuf + 2048;
    const _Float16* ws2h = wbuf + 6144;

    v16h a0 = frag_a(enchc + s0 * 32, lane, 32);
    v8f acc[4];
#pragma unroll
    for (int nt = 0; nt < 4; ++nt) { v8f z8 = {}; acc[nt] = WMMA(a0, frag_b(ws0h, lane, 0, nt, 4), z8); }
#pragma unroll
    for (int nt = 0; nt < 4; ++nt)
#pragma unroll
      for (int r = 0; r < 8; ++r)
        act[(s0 + r + 8 * hi) * AS + nt * 16 + col] = (_Float16)fmaxf(acc[nt][r], 0.f);

    v16h a1a = frag_a(act + s0 * AS, lane, AS);
    v16h a1b = frag_a(act + s0 * AS + 32, lane, AS);
#pragma unroll
    for (int nt = 0; nt < 4; ++nt) {
      v8f z8 = {};
      acc[nt] = WMMA(a1a, frag_b(ws1h, lane, 0, nt, 4), z8);
      acc[nt] = WMMA(a1b, frag_b(ws1h, lane, 1, nt, 4), acc[nt]);
    }
#pragma unroll
    for (int nt = 0; nt < 4; ++nt)
#pragma unroll
      for (int r = 0; r < 8; ++r)
        act[(s0 + r + 8 * hi) * AS + nt * 16 + col] = (_Float16)fmaxf(acc[nt][r], 0.f);

    v16h a2a = frag_a(act + s0 * AS, lane, AS);
    v16h a2b = frag_a(act + s0 * AS + 32, lane, AS);
    v8f z8 = {};
    v8f h2 = WMMA(a2a, frag_b(ws2h, lane, 0, 0, 1), z8);
    h2 = WMMA(a2b, frag_b(ws2h, lane, 1, 0, 1), h2);

    // sigma (col 0, keep-masked) + hc = [SH | geo | 0] reusing enchc
#pragma unroll
    for (int r = 0; r < 8; ++r) {
      int s = s0 + r + 8 * hi;
      if (col == 0) {
        sig[s] = (keep[s] != 0.f) ? h2[r] : 0.f;
        enchc[s * 32 + 31] = (_Float16)0.f;
      } else {
        enchc[s * 32 + 15 + col] = (_Float16)h2[r];
      }
      enchc[s * 32 + col] = (_Float16)shDir[col];
    }
  }
  __syncthreads();

  // ---- Phase 3: color-MLP weights into the same buffer (swizzled, zero-padded) ----
  for (int i = tid; i < 32 * 64; i += 256) {
    int k = i >> 6, n = i & 63;
    wbuf[bswz(k, n, 4)] = (k < 31) ? (_Float16)wc0[k * 64 + n] : (_Float16)0.f;
  }
  for (int i = tid; i < 64 * 64; i += 256) { int k = i >> 6, n = i & 63; wbuf[2048 + bswz(k, n, 4)] = (_Float16)wc1[i]; }
  for (int i = tid; i < 64 * 64; i += 256) { int k = i >> 6, n = i & 63; wbuf[6144 + bswz(k, n, 4)] = (_Float16)wc2[i]; }
  for (int i = tid; i < 64 * 16; i += 256) {
    int k = i >> 4, n = i & 15;
    wbuf[10240 + bswz(k, n, 1)] = (n < 3) ? (_Float16)wc3[k * 3 + n] : (_Float16)0.f;
  }
  __syncthreads();

  // ---- Phase 4: color MLP 32->64->64->64->3 ----
  {
    const _Float16* wc0h = wbuf;
    const _Float16* wc1h = wbuf + 2048;
    const _Float16* wc2h = wbuf + 6144;
    const _Float16* wc3h = wbuf + 10240;

    v16h ca = frag_a(enchc + s0 * 32, lane, 32);
    v8f acc[4];
#pragma unroll
    for (int nt = 0; nt < 4; ++nt) { v8f z8 = {}; acc[nt] = WMMA(ca, frag_b(wc0h, lane, 0, nt, 4), z8); }
#pragma unroll
    for (int nt = 0; nt < 4; ++nt)
#pragma unroll
      for (int r = 0; r < 8; ++r)
        act[(s0 + r + 8 * hi) * AS + nt * 16 + col] = (_Float16)fmaxf(acc[nt][r], 0.f);

    for (int layer = 0; layer < 2; ++layer) {
      const _Float16* w = (layer == 0) ? wc1h : wc2h;
      v16h aa = frag_a(act + s0 * AS, lane, AS);
      v16h ab = frag_a(act + s0 * AS + 32, lane, AS);
#pragma unroll
      for (int nt = 0; nt < 4; ++nt) {
        v8f z8 = {};
        acc[nt] = WMMA(aa, frag_b(w, lane, 0, nt, 4), z8);
        acc[nt] = WMMA(ab, frag_b(w, lane, 1, nt, 4), acc[nt]);
      }
#pragma unroll
      for (int nt = 0; nt < 4; ++nt)
#pragma unroll
        for (int r = 0; r < 8; ++r)
          act[(s0 + r + 8 * hi) * AS + nt * 16 + col] = (_Float16)fmaxf(acc[nt][r], 0.f);
    }

    v16h a3a = frag_a(act + s0 * AS, lane, AS);
    v16h a3b = frag_a(act + s0 * AS + 32, lane, AS);
    v8f z8 = {};
    v8f cc = WMMA(a3a, frag_b(wc3h, lane, 0, 0, 1), z8);
    cc = WMMA(a3b, frag_b(wc3h, lane, 1, 0, 1), cc);
#pragma unroll
    for (int r = 0; r < 8; ++r) {
      int s = s0 + r + 8 * hi;
      if (col < 3) colb[s * 3 + col] = cc[r];
    }
  }
  __syncthreads();

  // ---- Phase 5: per-sample alpha + sigmoid colors (parallel) ----
  if (tid < NS) {
    int s = tid;
    float dn = sqrtf(dx * dx + dy * dy + dz * dz);
    float dist = ((s < NS - 1) ? (FARZ - NEARZ) / (float)(NS - 1) : 1e10f) * dn;
    alphab[s] = 1.f - expf(-fmaxf(sig[s], 0.f) * dist);
#pragma unroll
    for (int c = 0; c < 3; ++c) colb[s * 3 + c] = 1.f / (1.f + expf(-colb[s * 3 + c]));
  }
  __syncthreads();

  // ---- Phase 6: serial transmittance scan (sum(probs) telescopes to 1+1e-6) ----
  if (tid == 0) {
    const float cden = 1.0f + 1e-6f;
    float T = 1.f, sumw = 0.f, c0 = 0.f, c1 = 0.f, c2 = 0.f, dep = 0.f, loss = 0.f;
    for (int s = 0; s < NS; ++s) {
      float a = alphab[s];
      float w = a * T;
      float z = NEARZ + (FARZ - NEARZ) * (float)s / (float)(NS - 1);
      c0 += w * colb[s * 3 + 0];
      c1 += w * colb[s * 3 + 1];
      c2 += w * colb[s * 3 + 2];
      dep += w * z;
      float p = w / cden;
      loss -= p * logf(fmaxf(p, 1e-37f));
      sumw += w;
      T *= (1.f - a + 1e-10f);
    }
    float plast = (1.f - sumw + 1e-6f) / cden;
    loss -= plast * logf(fmaxf(plast, 1e-37f));
    out[ray * 3 + 0] = c0;
    out[ray * 3 + 1] = c1;
    out[ray * 3 + 2] = c2;
    out[n_rays * 3 + ray] = dep;
    out[n_rays * 4 + ray] = loss;
  }
}

extern "C" void kernel_launch(void* const* d_in, const int* in_sizes, int n_in,
                              void* d_out, int out_size, void* d_ws, size_t ws_size,
                              hipStream_t stream) {
  const float* rays = (const float*)d_in[0];
  const float* tabs = (const float*)d_in[1];
  const float* ws0  = (const float*)d_in[2];
  const float* ws1  = (const float*)d_in[3];
  const float* ws2  = (const float*)d_in[4];
  const float* wc0  = (const float*)d_in[5];
  const float* wc1  = (const float*)d_in[6];
  const float* wc2  = (const float*)d_in[7];
  const float* wc3  = (const float*)d_in[8];
  float* out = (float*)d_out;
  int n_rays = in_sizes[0] / 6;
  nerf_fused_wmma<<<n_rays, 256, 0, stream>>>(rays, tabs, ws0, ws1, ws2,
                                              wc0, wc1, wc2, wc3, out, n_rays);
}
